// ASGFormer_41618233098812
// MI455X (gfx1250) — matrix operand order repaired and checked by reading, not compiled, not run on hardware
//
#include <hip/hip_runtime.h>

// ---------------------------------------------------------------------------
// ASGFormer layer for MI455X (gfx1250, wave32, WMMA).
// prep(weightsT->bf16) -> init -> fused feat GEMM+LN (WMMA) ->
// fused edge kernel (3 WMMA GEMMs + 2 LNs + score + seg-max) ->
// softmax pass -> weighted scatter-add -> output LN.
// ---------------------------------------------------------------------------

constexpr int NN = 30000;
constexpr int NE = 480000;
constexpr int C  = 128;

#define AST 136   // LDS stride (ushorts) for bf16 tiles: 272B rows, no bank clash
#define FST 132   // LDS stride (floats) for f32 tiles: 528B rows (16B aligned)

typedef __attribute__((ext_vector_type(16))) __bf16        v16bf;
typedef __attribute__((ext_vector_type(8)))  __bf16        v8bf;
typedef __attribute__((ext_vector_type(8)))  float         v8f;
typedef __attribute__((ext_vector_type(4)))  float         v4f;
typedef __attribute__((ext_vector_type(8)))  unsigned short v8us;
typedef __attribute__((ext_vector_type(4)))  unsigned short v4us;

__device__ __forceinline__ unsigned short f2bf(float f) {
  unsigned int u = __float_as_uint(f);
  unsigned int r = u + 0x7FFFu + ((u >> 16) & 1u);   // round-to-nearest-even
  return (unsigned short)(r >> 16);
}
__device__ __forceinline__ float bf2f(unsigned short h) {
  return __uint_as_float(((unsigned int)h) << 16);
}

// A-fragment (16x32 bf16) from an LDS tile, contiguous K order (same as load_b_frag):
// lane L<16 -> row M=L, K = {k0..k0+15}; L>=16 -> +16 on K base.
__device__ __forceinline__ v16bf load_a_frag(const unsigned short* lds, int k0, int lane) {
  const int r    = lane & 15;
  const int base = (lane >> 4) << 4;
  union { v16bf v; v8bf h[2]; } u;
  u.h[0] = *(const v8bf*)(lds + r * AST + k0 + base);
  u.h[1] = *(const v8bf*)(lds + r * AST + k0 + 8 + base);
  return u.v;
}

// B-fragment (32x16 bf16) from column-major bf16 weights wB[n][k]:
// lane holds column n = ncol+(lane&15), 16 contiguous K starting at k0+16*(lane>>4).
__device__ __forceinline__ v16bf load_b_frag(const unsigned short* wB, int ncol, int k0, int lane) {
  const int n  = ncol + (lane & 15);
  const int kb = k0 + ((lane >> 4) << 4);
  return *(const v16bf*)(wB + (size_t)n * C + kb);
}

__device__ __forceinline__ v8f wmma_bf16(v16bf a, v16bf b, v8f c) {
  return __builtin_amdgcn_wmma_f32_16x16x32_bf16(false, a, false, b, (short)0, c, false, false);
}

__device__ __forceinline__ float wave_allreduce_sum(float v) {
#pragma unroll
  for (int off = 16; off >= 1; off >>= 1) v += __shfl_xor(v, off, 32);
  return v;
}

__device__ __forceinline__ void atomicMaxFloat(float* addr, float val) {
  if (val >= 0.f) atomicMax((int*)addr, __float_as_int(val));
  else            atomicMin((unsigned int*)addr, (unsigned int)__float_as_int(val));
}

// ---------------------------------------------------------------------------
// Prep: transpose 4 [128x128] fp32 weight matrices to column-major bf16.
// grid (64,4) x 256
__global__ __launch_bounds__(256) void k_prep(
    const float* __restrict__ feat_w, const float* __restrict__ wf_w,
    const float* __restrict__ q_w,    const float* __restrict__ k_w,
    unsigned short* __restrict__ featB, unsigned short* __restrict__ wfB,
    unsigned short* __restrict__ qB,    unsigned short* __restrict__ kB) {
  int idx = blockIdx.x * 256 + threadIdx.x;   // 0..16383
  int k = idx >> 7, n = idx & 127;
  const float* src; unsigned short* dst;
  switch (blockIdx.y) {
    case 0:  src = feat_w; dst = featB; break;
    case 1:  src = wf_w;   dst = wfB;   break;   // first 128 rows of [131x128]
    case 2:  src = q_w;    dst = qB;    break;
    default: src = k_w;    dst = kB;    break;
  }
  dst[n * C + k] = f2bf(src[k * C + n]);
}

// grid 3750 x 256: agg zeroed with b128 stores, segmax=-inf, denom=0
__global__ __launch_bounds__(256) void k_init(float* __restrict__ agg,
                                              float* __restrict__ segmax,
                                              float* __restrict__ denom) {
  int i = blockIdx.x * 256 + threadIdx.x;     // 0..959999
  v4f z = {};
  *(v4f*)(agg + (size_t)i * 4) = z;
  if (i < NN) { segmax[i] = __int_as_float(0xFF800000); denom[i] = 0.f; }
}

// Fused feature GEMM + LN: featsBf = bf16(LN(relu(x @ feat_w + feat_b)))
// grid 1875 blocks (16 nodes each), 8 waves.
__global__ __launch_bounds__(256) void k_feat(const float* __restrict__ x,
                                              const unsigned short* __restrict__ featB,
                                              const float* __restrict__ fb,
                                              const float* __restrict__ fg,
                                              const float* __restrict__ fbt,
                                              unsigned short* __restrict__ featsBf) {
  __shared__ __align__(16) unsigned short aL[16 * AST];
  __shared__ __align__(16) float WL[16 * FST];
  const int t = threadIdx.x;
  const int tile_m = blockIdx.x;
  {
    int r = t >> 4, c0 = (t & 15) * 8;
    const v4f* xr = (const v4f*)(x + (size_t)(tile_m * 16 + r) * C + c0);
    v4f x0 = xr[0], x1 = xr[1];
    unsigned short* dstp = aL + r * AST + c0;
#pragma unroll
    for (int j = 0; j < 4; ++j) { dstp[j] = f2bf(x0[j]); dstp[4 + j] = f2bf(x1[j]); }
  }
  __syncthreads();
  const int wave = t >> 5, lane = t & 31;
  const int ncol = wave * 16;
  v8f acc = {};
#pragma unroll
  for (int k0 = 0; k0 < C; k0 += 32) {
    v16bf a = load_a_frag(aL, k0, lane);
    v16bf b = load_b_frag(featB, ncol, k0, lane);
    acc = wmma_bf16(a, b, acc);
  }
  const int n = ncol + (lane & 15);
  const int mBase = (lane >> 4) * 8;
  const float bias = fb[n];
#pragma unroll
  for (int v = 0; v < 8; ++v) WL[(mBase + v) * FST + n] = acc[v] + bias;
  __syncthreads();
  // LN: 2 rows per wave, lane owns 4 contiguous columns (b128 LDS loads)
  for (int rr = 2 * wave; rr <= 2 * wave + 1; ++rr) {
    v4f vals = *(const v4f*)(WL + rr * FST + lane * 4);
    float s = 0.f;
#pragma unroll
    for (int j = 0; j < 4; ++j) { vals[j] = vals[j] > 0.f ? vals[j] : 0.f; s += vals[j]; }
    s = wave_allreduce_sum(s);
    float mean = s * (1.f / C);
    float var = 0.f;
#pragma unroll
    for (int j = 0; j < 4; ++j) { float d = vals[j] - mean; var += d * d; }
    var = wave_allreduce_sum(var) * (1.f / C);
    float inv = rsqrtf(var + 1e-5f);
    v4f g = *(const v4f*)(fg + lane * 4);
    v4f b = *(const v4f*)(fbt + lane * 4);
    v4us h;
#pragma unroll
    for (int j = 0; j < 4; ++j) h[j] = f2bf((vals[j] - mean) * inv * g[j] + b[j]);
    *(v4us*)(featsBf + (size_t)(tile_m * 16 + rr) * C + lane * 4) = h;
  }
}

// ---------------------------------------------------------------------------
// Fused edge kernel: 16 edges/block, 8 waves. Three WMMA GEMMs + 2 LNs + score.
__global__ __launch_bounds__(256) void k_edge(
    const unsigned short* __restrict__ featsBf, const float* __restrict__ pos,
    const long long* __restrict__ ei,
    const unsigned short* __restrict__ wfB, const float* __restrict__ wf_w,
    const float* __restrict__ wf_b, const float* __restrict__ wf_g,
    const float* __restrict__ wf_bt,
    const unsigned short* __restrict__ qB, const float* __restrict__ q_b,
    const unsigned short* __restrict__ kB, const float* __restrict__ k_b,
    const float* __restrict__ pe_w, const float* __restrict__ pe_b,
    const float* __restrict__ pe_g, const float* __restrict__ pe_bt,
    unsigned short* __restrict__ Wout, float* __restrict__ scoreOut,
    float* __restrict__ segmax) {
  __shared__ int sSrc[16], sDst[16];
  __shared__ float dpL[16][4];
  __shared__ __align__(16) unsigned short dfL[16 * AST];   // bf16 delta_f
  __shared__ __align__(16) unsigned short xiL[16 * AST];   // bf16 x_i
  __shared__ __align__(16) unsigned short wbL[16 * AST];   // bf16 W_ij
  __shared__ __align__(16) float WL[16 * FST];
  __shared__ __align__(16) float PL[16 * FST];
  __shared__ __align__(16) float QL[16 * FST];
  __shared__ __align__(16) float KL[16 * FST];

  const int t = threadIdx.x;
  const int wave = t >> 5, lane = t & 31;
  const long long e0 = (long long)blockIdx.x * 16;
  const int ncol = wave * 16;
  const int n = ncol + (lane & 15);
  const int mBase = (lane >> 4) * 8;

  // warm L0 with this wave's weight column slices (global_prefetch_b8)
  __builtin_prefetch(wfB + (size_t)n * C, 0, 3);
  __builtin_prefetch(qB + (size_t)n * C, 0, 3);
  __builtin_prefetch(kB + (size_t)n * C, 0, 3);

  if (t < 16) {
    sSrc[t] = (int)ei[e0 + t];
    sDst[t] = (int)ei[(long long)NE + e0 + t];
  }
  __syncthreads();
  if (t < 48) {
    int r = t / 3, j = t % 3;
    dpL[r][j] = pos[sDst[r] * 3 + j] - pos[sSrc[r] * 3 + j];
  }
  {
    int r = t >> 4, c0 = (t & 15) * 8;
    v8us us = *(const v8us*)(featsBf + (size_t)sSrc[r] * C + c0);  // x_j chunk
    v8us ud = *(const v8us*)(featsBf + (size_t)sDst[r] * C + c0);  // x_i chunk
    *(v8us*)(xiL + r * AST + c0) = ud;
    v8us df;
#pragma unroll
    for (int j = 0; j < 8; ++j) df[j] = f2bf(bf2f(ud[j]) - bf2f(us[j]));
    *(v8us*)(dfL + r * AST + c0) = df;
  }
  __syncthreads();

  // GEMM1: Wraw = delta_f @ wf_w[0:128] + delta_p @ wf_w[128:131] + wf_b
  {
    v8f acc = {};
#pragma unroll
    for (int k0 = 0; k0 < C; k0 += 32) {
      v16bf a = load_a_frag(dfL, k0, lane);
      v16bf b = load_b_frag(wfB, ncol, k0, lane);
      acc = wmma_bf16(a, b, acc);
    }
    float w0 = wf_w[128 * C + n], w1 = wf_w[129 * C + n], w2 = wf_w[130 * C + n];
    float bias = wf_b[n];
#pragma unroll
    for (int v = 0; v < 8; ++v) {
      int m = mBase + v;
      WL[m * FST + n] = acc[v] + dpL[m][0] * w0 + dpL[m][1] * w1 + dpL[m][2] * w2 + bias;
    }
  }
  __syncthreads();

  // W_ij = LN(relu(Wraw)) -> WL (f32) + wbL (bf16).  2 rows/wave, 4 contig cols/lane.
  for (int rr = 2 * wave; rr <= 2 * wave + 1; ++rr) {
    v4f vals = *(const v4f*)(WL + rr * FST + lane * 4);
    float s = 0.f;
#pragma unroll
    for (int j = 0; j < 4; ++j) { vals[j] = vals[j] > 0.f ? vals[j] : 0.f; s += vals[j]; }
    s = wave_allreduce_sum(s);
    float mean = s * (1.f / C);
    float var = 0.f;
#pragma unroll
    for (int j = 0; j < 4; ++j) { float d = vals[j] - mean; var += d * d; }
    var = wave_allreduce_sum(var) * (1.f / C);
    float inv = rsqrtf(var + 1e-5f);
    v4f g = *(const v4f*)(wf_g + lane * 4);
    v4f b = *(const v4f*)(wf_bt + lane * 4);
    v4f o; v4us h;
#pragma unroll
    for (int j = 0; j < 4; ++j) { o[j] = (vals[j] - mean) * inv * g[j] + b[j]; h[j] = f2bf(o[j]); }
    *(v4f*)(WL + rr * FST + lane * 4) = o;
    *(v4us*)(wbL + rr * AST + lane * 4) = h;
  }
  // pos_emb = LN(relu(delta_p @ pe_w + pe_b)) -> PL
  for (int rr = 2 * wave; rr <= 2 * wave + 1; ++rr) {
    float d0 = dpL[rr][0], d1 = dpL[rr][1], d2 = dpL[rr][2];
    v4f w0 = *(const v4f*)(pe_w + lane * 4);
    v4f w1 = *(const v4f*)(pe_w + C + lane * 4);
    v4f w2 = *(const v4f*)(pe_w + 2 * C + lane * 4);
    v4f pb = *(const v4f*)(pe_b + lane * 4);
    v4f vals; float s = 0.f;
#pragma unroll
    for (int j = 0; j < 4; ++j) {
      float f = d0 * w0[j] + d1 * w1[j] + d2 * w2[j] + pb[j];
      f = f > 0.f ? f : 0.f;
      vals[j] = f; s += f;
    }
    s = wave_allreduce_sum(s);
    float mean = s * (1.f / C);
    float var = 0.f;
#pragma unroll
    for (int j = 0; j < 4; ++j) { float d = vals[j] - mean; var += d * d; }
    var = wave_allreduce_sum(var) * (1.f / C);
    float inv = rsqrtf(var + 1e-5f);
    v4f g = *(const v4f*)(pe_g + lane * 4);
    v4f b = *(const v4f*)(pe_bt + lane * 4);
    v4f o;
#pragma unroll
    for (int j = 0; j < 4; ++j) o[j] = (vals[j] - mean) * inv * g[j] + b[j];
    *(v4f*)(PL + rr * FST + lane * 4) = o;
  }
  __syncthreads();

  // spill W_ij bf16 to global: one b128 store per thread from LDS
  {
    int r = t >> 4, c0 = (t & 15) * 8;
    *(v8us*)(Wout + (size_t)(e0 + r) * C + c0) = *(const v8us*)(wbL + r * AST + c0);
  }

  // GEMM2: query = x_i @ q_w + q_b + pos_emb ; GEMM3: key = W_ij @ k_w + k_b
  {
    v8f qa = {};
#pragma unroll
    for (int k0 = 0; k0 < C; k0 += 32)
      qa = wmma_bf16(load_a_frag(xiL, k0, lane), load_b_frag(qB, ncol, k0, lane), qa);
    v8f ka = {};
#pragma unroll
    for (int k0 = 0; k0 < C; k0 += 32)
      ka = wmma_bf16(load_a_frag(wbL, k0, lane), load_b_frag(kB, ncol, k0, lane), ka);
    float qb_ = q_b[n], kb_ = k_b[n];
#pragma unroll
    for (int v = 0; v < 8; ++v) {
      int m = mBase + v;
      QL[m * FST + n] = qa[v] + qb_ + PL[m * FST + n];
      KL[m * FST + n] = ka[v] + kb_;
    }
  }
  __syncthreads();

  // score = dot(query,key)/sqrt(128); seg-max via ordered-int atomic
  for (int rr = 2 * wave; rr <= 2 * wave + 1; ++rr) {
    v4f q = *(const v4f*)(QL + rr * FST + lane * 4);
    v4f k = *(const v4f*)(KL + rr * FST + lane * 4);
    float s = q[0] * k[0] + q[1] * k[1] + q[2] * k[2] + q[3] * k[3];
    s = wave_allreduce_sum(s) * 0.08838834764831845f;
    if (lane == 0) {
      scoreOut[e0 + rr] = s;
      atomicMaxFloat(segmax + sDst[rr], s);
    }
  }
}

// exp(score - max[dst]) and denom accumulation. grid 1875 x 256
__global__ __launch_bounds__(256) void k_softmax(const long long* __restrict__ ei,
                                                 float* __restrict__ score,
                                                 const float* __restrict__ segmax,
                                                 float* __restrict__ denom) {
  int e = blockIdx.x * 256 + threadIdx.x;
  int d = (int)ei[(long long)NE + e];
  float v = __expf(score[e] - segmax[d]);
  score[e] = v;
  atomicAdd(&denom[d], v);
}

// agg[dst] += attn * W_ij.  grid 240000 x 256 (2 edges/block, 128 thr/edge)
__global__ __launch_bounds__(256) void k_agg(const long long* __restrict__ ei,
                                             const float* __restrict__ score,
                                             const float* __restrict__ denom,
                                             const unsigned short* __restrict__ W,
                                             float* __restrict__ agg) {
  int t = threadIdx.x;
  long long e = (long long)blockIdx.x * 2 + (t >> 7);
  int c = t & 127;
  int d = (int)ei[(long long)NE + e];
  float attn = score[e] / (denom[d] + 1e-16f);
  atomicAdd(&agg[(size_t)d * C + c], attn * bf2f(W[(size_t)e * C + c]));
}

// out = LN(agg + x).  grid 3750 x 256 (8 rows/block)
__global__ __launch_bounds__(256) void k_out(const float* __restrict__ agg,
                                             const float* __restrict__ x,
                                             const float* __restrict__ g,
                                             const float* __restrict__ bt,
                                             float* __restrict__ out) {
  const int wave = threadIdx.x >> 5, lane = threadIdx.x & 31;
  const int row = blockIdx.x * 8 + wave;
  v4f a = *(const v4f*)(agg + (size_t)row * C + lane * 4);
  v4f xv = *(const v4f*)(x + (size_t)row * C + lane * 4);
  v4f vals; float s = 0.f;
#pragma unroll
  for (int j = 0; j < 4; ++j) { vals[j] = a[j] + xv[j]; s += vals[j]; }
  s = wave_allreduce_sum(s);
  float mean = s * (1.f / C);
  float var = 0.f;
#pragma unroll
  for (int j = 0; j < 4; ++j) { float d = vals[j] - mean; var += d * d; }
  var = wave_allreduce_sum(var) * (1.f / C);
  float inv = rsqrtf(var + 1e-5f);
  v4f gv = *(const v4f*)(g + lane * 4);
  v4f bv = *(const v4f*)(bt + lane * 4);
  v4f o;
#pragma unroll
  for (int j = 0; j < 4; ++j) o[j] = (vals[j] - mean) * inv * gv[j] + bv[j];
  *(v4f*)(out + (size_t)row * C + lane * 4) = o;
}

// ---------------------------------------------------------------------------
extern "C" void kernel_launch(void* const* d_in, const int* in_sizes, int n_in,
                              void* d_out, int out_size, void* d_ws, size_t ws_size,
                              hipStream_t stream) {
  (void)in_sizes; (void)n_in; (void)out_size; (void)ws_size;
  const float*     x      = (const float*)d_in[0];
  const float*     pos    = (const float*)d_in[1];
  const long long* ei     = (const long long*)d_in[2];
  const float*     feat_w = (const float*)d_in[3];
  const float*     feat_b = (const float*)d_in[4];
  const float*     feat_g = (const float*)d_in[5];
  const float*     feat_bt= (const float*)d_in[6];
  const float*     wf_w   = (const float*)d_in[7];
  const float*     wf_b   = (const float*)d_in[8];
  const float*     wf_g   = (const float*)d_in[9];
  const float*     wf_bt  = (const float*)d_in[10];
  const float*     q_w    = (const float*)d_in[11];
  const float*     q_b    = (const float*)d_in[12];
  const float*     k_w    = (const float*)d_in[13];
  const float*     k_b    = (const float*)d_in[14];
  const float*     pe_w   = (const float*)d_in[15];
  const float*     pe_b   = (const float*)d_in[16];
  const float*     pe_g   = (const float*)d_in[17];
  const float*     pe_bt  = (const float*)d_in[18];
  const float*     out_g  = (const float*)d_in[19];
  const float*     out_bt = (const float*)d_in[20];

  char* w = (char*)d_ws;
  auto alloc = [&](size_t bytes) -> char* {
    char* p = w; w += (bytes + 255) & ~size_t(255); return p;
  };
  unsigned short* featB   = (unsigned short*)alloc((size_t)C * C * 2);
  unsigned short* wfB     = (unsigned short*)alloc((size_t)C * C * 2);
  unsigned short* qB      = (unsigned short*)alloc((size_t)C * C * 2);
  unsigned short* kB      = (unsigned short*)alloc((size_t)C * C * 2);
  unsigned short* featsBf = (unsigned short*)alloc((size_t)NN * C * 2);
  unsigned short* Wout    = (unsigned short*)alloc((size_t)NE * C * 2);
  float*          score   = (float*)alloc((size_t)NE * 4);
  float*          segmax  = (float*)alloc((size_t)NN * 4);
  float*          denom   = (float*)alloc((size_t)NN * 4);
  float*          agg     = (float*)alloc((size_t)NN * C * 4);

  k_prep<<<dim3(64, 4), 256, 0, stream>>>(feat_w, wf_w, q_w, k_w, featB, wfB, qB, kB);
  k_init<<<NN * C / 1024, 256, 0, stream>>>(agg, segmax, denom);
  k_feat<<<NN / 16, 256, 0, stream>>>(x, featB, feat_b, feat_g, feat_bt, featsBf);
  k_edge<<<NE / 16, 256, 0, stream>>>(featsBf, pos, ei,
                                      wfB, wf_w, wf_b, wf_g, wf_bt,
                                      qB, q_b, kB, k_b,
                                      pe_w, pe_b, pe_g, pe_bt,
                                      Wout, score, segmax);
  k_softmax<<<NE / 256, 256, 0, stream>>>(ei, score, segmax, denom);
  k_agg<<<NE / 2, 256, 0, stream>>>(ei, score, denom, Wout, agg);
  k_out<<<NN / 8, 256, 0, stream>>>(agg, x, out_g, out_bt, (float*)d_out);
}
